// MMDLoss_27968827031704
// MI455X (gfx1250) — compile-verified, matrix operand order checked by reading, not run on hardware
//
#include <hip/hip_runtime.h>
#include <hip/hip_bf16.h>

// MMD loss with RBF kernel:  loss = mean(Kxx) + mean(Kyy) - 2*mean(Kxy)
// Phase 1: one-time fp32 -> (bf16 hi, bf16 lo) split into workspace + row norms.
// Phase 2: Gram tiles via bf16 WMMA (hi*hi + hi*lo + lo*hi ~ fp32 accuracy),
//          exp + mean fused in the epilogue, one scalar atomic per block.
// Hot-loop addressing: uniform (SGPR) base advanced by k0 + per-lane 32-bit
// offset -> GVS-form global_load_b128, no vector address math, no WMMA->VALU
// WAR hazard nops.

typedef __attribute__((ext_vector_type(16))) __bf16 v16bf;
typedef __attribute__((ext_vector_type(8)))  __bf16 v8bf;
typedef __attribute__((ext_vector_type(4)))  __bf16 v4bf;
typedef __attribute__((ext_vector_type(8)))  float  v8f;

#define DIM   1024
#define NROWS 4096

__global__ void mmd_zero(float* __restrict__ out) { out[0] = 0.0f; }

// One-time split: x = hi + lo (both bf16, RNE). 4 elements per thread, both tensors.
__global__ void __launch_bounds__(256)
mmd_split(const float* __restrict__ X, const float* __restrict__ Y,
          __bf16* __restrict__ Xhi, __bf16* __restrict__ Xlo,
          __bf16* __restrict__ Yhi, __bf16* __restrict__ Ylo) {
  const size_t i4 = ((size_t)blockIdx.x * 256 + threadIdx.x) * 4;
  float4 x = *reinterpret_cast<const float4*>(X + i4);
  float4 y = *reinterpret_cast<const float4*>(Y + i4);
  float fx[4] = {x.x, x.y, x.z, x.w};
  float fy[4] = {y.x, y.y, y.z, y.w};
  v4bf xh, xl, yh, yl;
#pragma unroll
  for (int i = 0; i < 4; ++i) {
    __bf16 h = (__bf16)fx[i];
    xh[i] = h;
    xl[i] = (__bf16)(fx[i] - (float)h);
    __bf16 g = (__bf16)fy[i];
    yh[i] = g;
    yl[i] = (__bf16)(fy[i] - (float)g);
  }
  *reinterpret_cast<v4bf*>(Xhi + i4) = xh;
  *reinterpret_cast<v4bf*>(Xlo + i4) = xl;
  *reinterpret_cast<v4bf*>(Yhi + i4) = yh;
  *reinterpret_cast<v4bf*>(Ylo + i4) = yl;
}

// One block per row (row<4096 -> X, else Y): squared L2 norm.
__global__ void __launch_bounds__(256)
mmd_rownorms(const float* __restrict__ X, const float* __restrict__ Y,
             float* __restrict__ xsq, float* __restrict__ ysq) {
  const int row = blockIdx.x;
  const float* src;
  float* dst;
  if (row < NROWS) { src = X + (size_t)row * DIM;           dst = xsq + row; }
  else             { src = Y + (size_t)(row - NROWS) * DIM; dst = ysq + (row - NROWS); }
  float s = 0.0f;
  for (int i = threadIdx.x; i < DIM; i += 256) { float v = src[i]; s += v * v; }
#pragma unroll
  for (int off = 16; off > 0; off >>= 1) s += __shfl_xor(s, off, 32);
  __shared__ float wsum[8];
  const int lane = threadIdx.x & 31, wid = threadIdx.x >> 5;
  if (lane == 0) wsum[wid] = s;
  __syncthreads();
  if (threadIdx.x == 0) {
    float t = 0.0f;
#pragma unroll
    for (int i = 0; i < 8; ++i) t += wsum[i];
    *dst = t;
  }
}

// 16x32 bf16 fragment load from (uniform base + k0) with per-lane element
// offset; two 16-byte loads -> GVS addressing (saddr + v_offset [+ imm]).
__device__ __forceinline__ v16bf load_frag_off(const __bf16* __restrict__ sk, int off) {
  v8bf c0 = *reinterpret_cast<const v8bf*>(sk + off);
  v8bf c1 = *reinterpret_cast<const v8bf*>(sk + off + 16);
  return __builtin_shufflevector(c0, c1, 0, 1, 2, 3, 4, 5, 6, 7,
                                 8, 9, 10, 11, 12, 13, 14, 15);
}

// blockIdx.z: 0 = XX, 1 = YY, 2 = XY.  128 threads = 4 wave32, 2x2 wave grid,
// 128x128 block tile, 64x64 per wave as 4x4 grid of 16x16 WMMA accumulators.
__global__ void __launch_bounds__(128)
mmd_gram(const __bf16* __restrict__ Xhi, const __bf16* __restrict__ Xlo,
         const __bf16* __restrict__ Yhi, const __bf16* __restrict__ Ylo,
         const float* __restrict__ xsq, const float* __restrict__ ysq,
         float* __restrict__ out) {
  const int z = blockIdx.z;
  const __bf16 *AH, *AL, *BH, *BL;
  const float *Aq, *Bq;
  float w;
  if (z == 0)      { AH = Xhi; AL = Xlo; BH = Xhi; BL = Xlo; Aq = xsq; Bq = xsq; w =  1.0f; }
  else if (z == 1) { AH = Yhi; AL = Ylo; BH = Yhi; BL = Ylo; Aq = ysq; Bq = ysq; w =  1.0f; }
  else             { AH = Xhi; AL = Xlo; BH = Yhi; BL = Ylo; Aq = xsq; Bq = ysq; w = -2.0f; }

  const int lane = threadIdx.x & 31;
  const int wid  = threadIdx.x >> 5;
  const int m0 = blockIdx.y * 128 + (wid >> 1) * 64;   // rows of A
  const int n0 = blockIdx.x * 128 + (wid &  1) * 64;   // rows of B (cols of Gram)

  // Per-lane fragment offsets (elements), fixed for the whole k-loop.
  // CDNA5 16-bit A-matrix 16x32 layout:
  //   lanes 0-15  : row M = lane,    K = k0+{0..7, 16..23}
  //   lanes 16-31 : row M = lane-16, K = k0+{8..15, 24..31}
  const int rsel = lane & 15;
  const int kh   = (lane >> 4) << 3;                   // 0 or 8
  int offA[4], offB[4];
#pragma unroll
  for (int t = 0; t < 4; ++t) {
    offA[t] = (m0 + t * 16 + rsel) * DIM + kh;
    offB[t] = (n0 + t * 16 + rsel) * DIM + kh;
  }

  v8f acc[4][4];
  v8f zf = {};
#pragma unroll
  for (int i = 0; i < 4; ++i)
#pragma unroll
    for (int j = 0; j < 4; ++j) acc[i][j] = zf;

#pragma unroll 2
  for (int k0 = 0; k0 < DIM; k0 += 32) {
    const __bf16* AHk = AH + k0;   // uniform: folds to SALU base update
    const __bf16* ALk = AL + k0;
    const __bf16* BHk = BH + k0;
    const __bf16* BLk = BL + k0;
    v16bf ahi[4], alo[4], bhi[4], blo[4];
#pragma unroll
    for (int t = 0; t < 4; ++t) {
      ahi[t] = load_frag_off(AHk, offA[t]);
      alo[t] = load_frag_off(ALk, offA[t]);
      bhi[t] = load_frag_off(BHk, offB[t]);
      blo[t] = load_frag_off(BLk, offB[t]);
    }
#pragma unroll
    for (int i = 0; i < 4; ++i)
#pragma unroll
      for (int j = 0; j < 4; ++j) {
        v8f c = acc[i][j];
        c = __builtin_amdgcn_wmma_f32_16x16x32_bf16(false, ahi[i], false, bhi[j],
                                                    (short)0, c, false, false);
        c = __builtin_amdgcn_wmma_f32_16x16x32_bf16(false, ahi[i], false, blo[j],
                                                    (short)0, c, false, false);
        c = __builtin_amdgcn_wmma_f32_16x16x32_bf16(false, alo[i], false, bhi[j],
                                                    (short)0, c, false, false);
        acc[i][j] = c;
      }
  }

  // Epilogue: d = ||x||^2 + ||y||^2 - 2*cross ; accumulate exp(-d/32).
  // C/D layout: VGPR r -> M = r (lanes 0-15) or r+8 (lanes 16-31); N = lane&15.
  const int moff = (lane >= 16) ? 8 : 0;
  float asv[4][8];
#pragma unroll
  for (int i = 0; i < 4; ++i)
#pragma unroll
    for (int r = 0; r < 8; ++r)
      asv[i][r] = Aq[m0 + i * 16 + moff + r];
  float bsv[4];
#pragma unroll
  for (int j = 0; j < 4; ++j) bsv[j] = Bq[n0 + j * 16 + rsel];

  float lsum = 0.0f;
#pragma unroll
  for (int i = 0; i < 4; ++i)
#pragma unroll
    for (int j = 0; j < 4; ++j)
#pragma unroll
      for (int r = 0; r < 8; ++r) {
        float d = asv[i][r] + bsv[j] - 2.0f * acc[i][j][r];
        lsum += __expf(d * (-1.0f / 32.0f));
      }

  // wave32 reduce, then cross-wave via LDS, one atomic per block
#pragma unroll
  for (int off = 16; off > 0; off >>= 1) lsum += __shfl_xor(lsum, off, 32);
  __shared__ float ssum[4];
  if (lane == 0) ssum[wid] = lsum;
  __syncthreads();
  if (threadIdx.x == 0) {
    float s = ssum[0] + ssum[1] + ssum[2] + ssum[3];
    atomicAdd(out, w * s * (1.0f / ((float)NROWS * (float)NROWS)));
  }
}

extern "C" void kernel_launch(void* const* d_in, const int* in_sizes, int n_in,
                              void* d_out, int out_size, void* d_ws, size_t ws_size,
                              hipStream_t stream) {
  (void)in_sizes; (void)n_in; (void)out_size; (void)ws_size;
  const float* X = (const float*)d_in[0];   // [4096, 1024] fp32
  const float* Y = (const float*)d_in[1];   // [4096, 1024] fp32
  float* out = (float*)d_out;               // scalar fp32

  // Workspace layout: norms (32 KB) + four bf16 planes (4 x 8 MB = 32 MB)
  float*  xsq = (float*)d_ws;               // [4096]
  float*  ysq = xsq + NROWS;                // [4096]
  __bf16* Xhi = (__bf16*)(ysq + NROWS);     // [4096*1024]
  __bf16* Xlo = Xhi + (size_t)NROWS * DIM;
  __bf16* Yhi = Xlo + (size_t)NROWS * DIM;
  __bf16* Ylo = Yhi + (size_t)NROWS * DIM;

  mmd_zero<<<1, 1, 0, stream>>>(out);
  mmd_split<<<(NROWS * DIM) / (256 * 4), 256, 0, stream>>>(X, Y, Xhi, Xlo, Yhi, Ylo);
  mmd_rownorms<<<2 * NROWS, 256, 0, stream>>>(X, Y, xsq, ysq);

  dim3 grid(NROWS / 128, NROWS / 128, 3);   // 32 x 32 x {XX, YY, XY}
  mmd_gram<<<grid, 128, 0, stream>>>(Xhi, Xlo, Yhi, Ylo, xsq, ysq, out);
}
